// QuantumLayer_31190052503968
// MI455X (gfx1250) — compile-verified, matrix operand order checked by reading, not run on hardware
//
#include <hip/hip_runtime.h>

// ---------------------------------------------------------------------------
// 8-qubit statevector QNN layer for MI455X (gfx1250, wave32).
//   - one wave32 per batch element; full 256-amplitude state in VGPRs
//     (8 complex = 16 f32 per lane;  amp index i = lane*8 + j,
//      wires 0..4 <-> lane bits 16/8/4/2/1, wires 5..7 <-> j bits 4/2/1)
//   - H^8 folded into uniform init; RZ*RY*RZ*RY(angle) fused to one 2x2 gate
//   - cross-lane gates/CNOTs via ds_swizzle_b32 xor patterns
//   - epilogue probs(8x256) @ signs(256x8) via v_wmma_f32_16x16x4_f32,
//     K split over the 8 waves, reduced with LDS ds_add_f32
// ---------------------------------------------------------------------------

#define NQ 8
#define NL 3
#define WAVES_PER_WG 8
#define PADW 264  // 256 + 8 pad (stride%64==8 -> spreads LDS banks)

typedef __attribute__((ext_vector_type(2)))  float    v2f;
typedef __attribute__((ext_vector_type(8)))  float    v8f;
typedef __attribute__((ext_vector_type(16))) _Float16 v16h;

struct cplx { float x, y; };

// ds_swizzle_b32, group-of-32 mode: offset = {0, xor[14:10], or[9:5], and[4:0]}
template <int MASK>
__device__ __forceinline__ float swx(float v) {
  return __int_as_float(
      __builtin_amdgcn_ds_swizzle(__float_as_int(v), (MASK << 10) | 0x1F));
}

__device__ __forceinline__ float bcast(float v, int srcLane) {
  return __int_as_float(__builtin_amdgcn_readlane(__float_as_int(v), srcLane));
}

// 1-qubit gate on a lane-bit wire (xor mask MASK): pairs across lanes.
template <int MASK>
__device__ __forceinline__ void gate_lane(cplx* s, int lane, cplx u00, cplx u01,
                                          cplx u10, cplx u11) {
  const bool hi = (lane & MASK) != 0;
  cplx p, q;  // self / partner coefficients
  p.x = hi ? u11.x : u00.x;  p.y = hi ? u11.y : u00.y;
  q.x = hi ? u10.x : u01.x;  q.y = hi ? u10.y : u01.y;
#pragma unroll
  for (int j = 0; j < 8; ++j) {
    float ox = swx<MASK>(s[j].x);
    float oy = swx<MASK>(s[j].y);
    float nx = p.x * s[j].x - p.y * s[j].y + q.x * ox - q.y * oy;
    float ny = p.x * s[j].y + p.y * s[j].x + q.x * oy + q.y * ox;
    s[j].x = nx; s[j].y = ny;
  }
}

// 1-qubit gate on an in-lane wire (j-bit mask MJ): pure register math.
template <int MJ>
__device__ __forceinline__ void gate_j(cplx* s, cplx u00, cplx u01, cplx u10,
                                       cplx u11) {
#pragma unroll
  for (int j = 0; j < 8; ++j) {
    if ((j & MJ) == 0) {
      const int k = j | MJ;
      cplx a = s[j], b = s[k];
      s[j].x = u00.x * a.x - u00.y * a.y + u01.x * b.x - u01.y * b.y;
      s[j].y = u00.x * a.y + u00.y * a.x + u01.x * b.y + u01.y * b.x;
      s[k].x = u10.x * a.x - u10.y * a.y + u11.x * b.x - u11.y * b.y;
      s[k].y = u10.x * a.y + u10.y * a.x + u11.x * b.y + u11.y * b.x;
    }
  }
}

__device__ __forceinline__ void apply_gate(int wire, cplx* s, int lane,
                                           cplx u00, cplx u01, cplx u10, cplx u11) {
  // 'wire' is a constant after full unrolling -> dead branches eliminated.
  if      (wire == 0) gate_lane<16>(s, lane, u00, u01, u10, u11);
  else if (wire == 1) gate_lane<8>(s, lane, u00, u01, u10, u11);
  else if (wire == 2) gate_lane<4>(s, lane, u00, u01, u10, u11);
  else if (wire == 3) gate_lane<2>(s, lane, u00, u01, u10, u11);
  else if (wire == 4) gate_lane<1>(s, lane, u00, u01, u10, u11);
  else if (wire == 5) gate_j<4>(s, u00, u01, u10, u11);
  else if (wire == 6) gate_j<2>(s, u00, u01, u10, u11);
  else                gate_j<1>(s, u00, u01, u10, u11);
}

// CNOT: control lane-bit MC, target lane-bit MT.
template <int MC, int MT>
__device__ __forceinline__ void cnot_ll(cplx* s, int lane) {
  const bool c = (lane & MC) != 0;
#pragma unroll
  for (int j = 0; j < 8; ++j) {
    float ox = swx<MT>(s[j].x);
    float oy = swx<MT>(s[j].y);
    s[j].x = c ? ox : s[j].x;
    s[j].y = c ? oy : s[j].y;
  }
}

__global__ __launch_bounds__(256)
void qsim_wmma_kernel(const float* __restrict__ x, const float* __restrict__ w,
                      const float* __restrict__ iscale,
                      const float* __restrict__ oscale,
                      float* __restrict__ out, int Btot) {
  __shared__ float probs[WAVES_PER_WG][PADW];
  __shared__ float Cbuf[256];  // 16x16 f32 output tile (rows 8..15 / cols 8..15 unused)

  const int tid  = threadIdx.x;
  const int lane = tid & 31;
  const int wave = tid >> 5;
  const int batch  = blockIdx.x * WAVES_PER_WG + wave;
  const int bclamp = (batch < Btot) ? batch : (Btot - 1);

  Cbuf[tid] = 0.0f;

  // --- per-wave trig prep (all lanes busy) -------------------------------
  // lanes 0..23: trainable-rotation trig for (layer,wire) = (lane/8, lane%8)
  // lanes 24..31: data-angle trig for wire = lane&7
  const int widx = lane % 24;
  const float wa = w[widx * 3 + 0];
  const float wb = w[widx * 3 + 1];
  const float wc = w[widx * 3 + 2];
  float sb, cb, sp, cp, sm, cm;
  __sincosf(0.5f * wb, &sb, &cb);
  __sincosf(0.5f * (wa + wc), &sp, &cp);
  __sincosf(0.5f * (wa - wc), &sm, &cm);

  const int ai = lane & 7;
  const float ang = tanhf(x[bclamp * NQ + ai]) * iscale[ai];
  float sg_r, cg_r;
  __sincosf(0.5f * ang, &sg_r, &cg_r);

  // --- state = H^8 |0>: uniform 1/16 ------------------------------------
  cplx s[8];
#pragma unroll
  for (int j = 0; j < 8; ++j) { s[j].x = 0.0625f; s[j].y = 0.0f; }

#pragma unroll
  for (int l = 0; l < NL; ++l) {
    // fused U = RZ(c)*RY(b)*RZ(a)*RY(angle), one gate per wire
#pragma unroll
    for (int i = 0; i < NQ; ++i) {
      const int srcW = l * 8 + i;
      const float CB = bcast(cb, srcW), SB = bcast(sb, srcW);
      const float CP = bcast(cp, srcW), SP = bcast(sp, srcW);
      const float CM = bcast(cm, srcW), SM = bcast(sm, srcW);
      const float CG = bcast(cg_r, 24 + i), SG = bcast(sg_r, 24 + i);
      const float cbcp = CB * CP, cbsp = CB * SP;
      const float sbcm = SB * CM, sbsm = SB * SM;
      cplx u00, u01, u10, u11;
      u00.x =  cbcp * CG - sbcm * SG;  u00.y = -cbsp * CG - sbsm * SG;
      u01.x = -cbcp * SG - sbcm * CG;  u01.y =  cbsp * SG - sbsm * CG;
      u10.x =  sbcm * CG + cbcp * SG;  u10.y = -sbsm * CG + cbsp * SG;
      u11.x = -sbcm * SG + cbcp * CG;  u11.y =  sbsm * SG + cbsp * CG;
      apply_gate(i, s, lane, u00, u01, u10, u11);
    }
    // CNOT ring: (0,1)(1,2)(2,3)(3,4)(4,5)(5,6)(6,7)(7,0)
    cnot_ll<16, 8>(s, lane);
    cnot_ll<8, 4>(s, lane);
    cnot_ll<4, 2>(s, lane);
    cnot_ll<2, 1>(s, lane);
    {  // (4,5): control lane bit 1, target j bit 4 -> cndmask swap
      const bool c = (lane & 1) != 0;
#pragma unroll
      for (int j = 0; j < 4; ++j) {
        const int k = j | 4;
        cplx a = s[j], b = s[k];
        s[j].x = c ? b.x : a.x;  s[j].y = c ? b.y : a.y;
        s[k].x = c ? a.x : b.x;  s[k].y = c ? a.y : b.y;
      }
    }
    { cplx t = s[4]; s[4] = s[6]; s[6] = t;   // (5,6): static rename
      t = s[5]; s[5] = s[7]; s[7] = t; }
    { cplx t = s[2]; s[2] = s[3]; s[3] = t;   // (6,7): static rename
      t = s[6]; s[6] = s[7]; s[7] = t; }
#pragma unroll
    for (int j = 1; j < 8; j += 2) {          // (7,0): odd j swap lanes^16
      s[j].x = swx<16>(s[j].x);
      s[j].y = swx<16>(s[j].y);
    }
  }

  // --- probabilities to LDS ---------------------------------------------
#pragma unroll
  for (int j = 0; j < 8; ++j)
    probs[wave][lane * 8 + j] = s[j].x * s[j].x + s[j].y * s[j].y;
  __syncthreads();

  // --- epilogue GEMM: C[m][n] = sum_k probs[m][k] * sign_n(k) ------------
  // sign_n(k) = 1 - 2*((k >> (7-n)) & 1).  K split across the 8 waves.
  const int n    = lane & 15;
  const int row  = lane & 7;              // A rows 8..15 duplicate 0..7 (unused C rows)
  const int koff = (lane >> 4) << 1;      // f32 WMMA A/B: hi half-wave holds K+2,K+3
  const bool nv  = (n < 8);
  const int sh   = nv ? (7 - n) : 0;

#if __has_builtin(__builtin_amdgcn_wmma_f32_16x16x4_f32)
  {
    v8f acc = {0.f, 0.f, 0.f, 0.f, 0.f, 0.f, 0.f, 0.f};
#pragma unroll
    for (int kk = 0; kk < 32; kk += 4) {
      const int kb = wave * 32 + kk + koff;
      v2f a;
      a.x = probs[row][kb];
      a.y = probs[row][kb + 1];
      v2f bf;
      bf.x = nv ? (1.0f - 2.0f * (float)((kb >> sh) & 1)) : 0.0f;
      bf.y = nv ? (1.0f - 2.0f * (float)(((kb + 1) >> sh) & 1)) : 0.0f;
      acc = __builtin_amdgcn_wmma_f32_16x16x4_f32(false, a, false, bf,
                                                  (short)0, acc, false, false);
    }
    const int mbase = (lane >> 4) * 8;
#pragma unroll
    for (int r = 0; r < 8; ++r)
      atomicAdd(&Cbuf[(mbase + r) * 16 + n], acc[r]);
  }
#else
  {  // fallback: one v_wmma_f32_16x16x32_f16 per wave (codegen-confirmed builtin)
    v8f acc = {0.f, 0.f, 0.f, 0.f, 0.f, 0.f, 0.f, 0.f};
    const int kbase = wave * 32;
    v16h a, bm;
#pragma unroll
    for (int v = 0; v < 8; ++v) {
      const int k = kbase + ((v < 4) ? 2 * v : 16 + 2 * (v - 4)) +
                    ((lane >= 16) ? 8 : 0);
      a[2 * v]     = (_Float16)probs[row][k];
      a[2 * v + 1] = (_Float16)probs[row][k + 1];
      bm[2 * v]     = (_Float16)(nv ? (1.0f - 2.0f * (float)((k >> sh) & 1)) : 0.0f);
      bm[2 * v + 1] = (_Float16)(nv ? (1.0f - 2.0f * (float)(((k + 1) >> sh) & 1)) : 0.0f);
    }
    acc = __builtin_amdgcn_wmma_f32_16x16x32_f16(false, a, false, bm,
                                                 (short)0, acc, false, false);
    const int mbase = (lane >> 4) * 8;
#pragma unroll
    for (int r = 0; r < 8; ++r)
      atomicAdd(&Cbuf[(mbase + r) * 16 + n], acc[r]);
  }
#endif
  __syncthreads();

  // --- scaled writeout ---------------------------------------------------
  if (tid < 64) {
    const int m  = tid >> 3;
    const int nn = tid & 7;
    const int bb = blockIdx.x * WAVES_PER_WG + m;
    if (bb < Btot) out[bb * NQ + nn] = Cbuf[m * 16 + nn] * oscale[nn];
  }
}

extern "C" void kernel_launch(void* const* d_in, const int* in_sizes, int n_in,
                              void* d_out, int out_size, void* d_ws, size_t ws_size,
                              hipStream_t stream) {
  const float* x      = (const float*)d_in[0];  // (B, 8)
  const float* w      = (const float*)d_in[1];  // (3, 8, 3)
  const float* iscale = (const float*)d_in[2];  // (8,)
  const float* oscale = (const float*)d_in[3];  // (8,)
  float* out = (float*)d_out;                   // (B, 8)

  const int B = in_sizes[0] / NQ;
  const int blocks = (B + WAVES_PER_WG - 1) / WAVES_PER_WG;
  qsim_wmma_kernel<<<blocks, 256, 0, stream>>>(x, w, iscale, oscale, out, B);
}
